// GATShortRange_67095979098195
// MI455X (gfx1250) — matrix operand order repaired and checked by reading, not compiled, not run on hardware
//
#include <hip/hip_runtime.h>
#include <hip/hip_bf16.h>

#define BSZq 4
#define SEQq 2048
#define DDIM 128
#define HHq  4
#define EEq  65536
#define CCq  128
#define NNq  (BSZq * SEQq)        /* 8192  */
#define HCq  (HHq * CCq)          /* 512   */
#define ETOT (BSZq * EEq + NNq)   /* 270336 */

typedef __attribute__((ext_vector_type(16))) __bf16 v16bf;
typedef __attribute__((ext_vector_type(8)))  __bf16 v8bf;
typedef __attribute__((ext_vector_type(8)))  float  v8f;

__device__ __forceinline__ void atomicAddF32(float* p, float v) {
    __hip_atomic_fetch_add(p, v, __ATOMIC_RELAXED, __HIP_MEMORY_SCOPE_AGENT);
}

__device__ __forceinline__ void edge_decode(int t, const int* __restrict__ ei,
                                            int& s, int& d) {
    if (t < BSZq * EEq) {
        const int b   = t >> 16;          // EEq == 65536
        const int e   = t & (EEq - 1);
        const int off = b * SEQq;
        s = ei[e] + off;
        d = ei[EEq + e] + off;
    } else {
        s = d = t - BSZq * EEq;
    }
}

__device__ __forceinline__ v8bf cvt8(float4 a, float4 b) {
    v8bf o;
    o[0] = (__bf16)a.x; o[1] = (__bf16)a.y; o[2] = (__bf16)a.z; o[3] = (__bf16)a.w;
    o[4] = (__bf16)b.x; o[5] = (__bf16)b.y; o[6] = (__bf16)b.z; o[7] = (__bf16)b.w;
    return o;
}

// ------------------------------------------- f32 -> bf16 bulk convert ------
__global__ __launch_bounds__(256) void gat_cvt(const float* __restrict__ in,
                                               __bf16* __restrict__ out) {
    const int i = (blockIdx.x * 256 + threadIdx.x) * 8;
    float4 a = *(const float4*)(in + i);
    float4 b = *(const float4*)(in + i + 4);
    *(v8bf*)(out + i) = cvt8(a, b);
}

// ------------------------- (AGG + bias) -> bf16 convert for GEMM2 A ---------
__global__ __launch_bounds__(256) void gat_cvt_agg(const float* __restrict__ agg,
                                                   const float* __restrict__ bias,
                                                   __bf16* __restrict__ out) {
    const int i = (blockIdx.x * 256 + threadIdx.x) * 8;
    const int k = i & (HCq - 1);
    float4 a = *(const float4*)(agg + i);
    float4 b = *(const float4*)(agg + i + 4);
    float4 c = *(const float4*)(bias + k);
    float4 d = *(const float4*)(bias + k + 4);
    a.x += c.x; a.y += c.y; a.z += c.z; a.w += c.w;
    b.x += d.x; b.y += d.y; b.z += d.z; b.w += d.w;
    *(v8bf*)(out + i) = cvt8(a, b);
}

// ---------------------------------------------------------------- init ------
__global__ __launch_bounds__(256) void gat_init(float* __restrict__ agg,
                                                float* __restrict__ maxb,
                                                float* __restrict__ denom) {
    const int i = blockIdx.x * 256 + threadIdx.x;
    if (i < NNq * HCq) agg[i] = 0.0f;
    if (i < NNq * HHq) { maxb[i] = -__builtin_inff(); denom[i] = 0.0f; }
}

// ---------------------------------------- bf16 WMMA fragment assembly ------
__device__ __forceinline__ v16bf load_afrag(const __bf16* __restrict__ row, int ka) {
    v8bf lo8 = *(const v8bf*)(row + ka);        // K = ka .. ka+7
    v8bf hi8 = *(const v8bf*)(row + ka + 8);    // K = ka+8 .. ka+15
    return __builtin_shufflevector(lo8, hi8, 0, 1, 2, 3, 4, 5, 6, 7,
                                             8, 9, 10, 11, 12, 13, 14, 15);
}

// ------------------------------------------------------- GEMM1: x = s @ W^T -
// one 16x16 output tile per wave; 4 waves/block; K = 128 in 4 WMMA steps
__global__ __launch_bounds__(128) void gat_gemm1(const __bf16* __restrict__ S,
                                                 const __bf16* __restrict__ W,
                                                 float* __restrict__ X) {
    const int lane = threadIdx.x & 31;
    const int wave = threadIdx.x >> 5;
    const int lo   = lane & 15;
    const int hi   = lane >> 4;
    const int n0   = blockIdx.x << 4;
    const int m0   = (blockIdx.y << 6) + (wave << 4);
    const __bf16* __restrict__ srow = S + (size_t)(n0 + lo) * DDIM;
    const __bf16* __restrict__ wrow = W + (size_t)(m0 + lo) * DDIM;
    v8f acc = {};
#pragma unroll
    for (int kk = 0; kk < DDIM; kk += 32) {
        v16bf afrag = load_afrag(srow, kk + (hi << 4));
        v16bf bfrag = *(const v16bf*)(wrow + kk + (hi << 4));   // K = kb..kb+15
        acc = __builtin_amdgcn_wmma_f32_16x16x32_bf16(false, afrag, false, bfrag,
                                                      (short)0, acc, false, false);
    }
    const int col = m0 + lo;
    const int rb  = n0 + (hi << 3);
#pragma unroll
    for (int r = 0; r < 8; ++r)
        X[(size_t)(rb + r) * HCq + col] = acc[r];
}

// ------------------------------------------ per-(node,head) attention terms -
__global__ __launch_bounds__(256) void gat_attn_terms(const float* __restrict__ X,
                                                      const float* __restrict__ ASRC,
                                                      const float* __restrict__ ADST,
                                                      float* __restrict__ a_src,
                                                      float* __restrict__ a_dst) {
    const int gw   = (blockIdx.x * 256 + threadIdx.x) >> 5;  // one wave per (n,h)
    const int lane = threadIdx.x & 31;
    const int n = gw >> 2;
    const int h = gw & 3;
    const float4 xv = *(const float4*)(X + (size_t)n * HCq + h * CCq + lane * 4);
    const float4 sv = *(const float4*)(ASRC + h * CCq + lane * 4);
    const float4 dv = *(const float4*)(ADST + h * CCq + lane * 4);
    float s1 = xv.x * sv.x + xv.y * sv.y + xv.z * sv.z + xv.w * sv.w;
    float s2 = xv.x * dv.x + xv.y * dv.y + xv.z * dv.z + xv.w * dv.w;
#pragma unroll
    for (int off = 16; off > 0; off >>= 1) {
        s1 += __shfl_xor(s1, off, 32);
        s2 += __shfl_xor(s2, off, 32);
    }
    if (lane == 0) { a_src[gw] = s1; a_dst[gw] = s2; }
}

// ------------------------------------------------- edge pass 1: segment max -
__global__ __launch_bounds__(256) void gat_edge_max(const int* __restrict__ ei,
                                                    const float* __restrict__ a_src,
                                                    const float* __restrict__ a_dst,
                                                    float* __restrict__ maxb) {
    const int t = blockIdx.x * 256 + threadIdx.x;
    if (t >= ETOT) return;
    int s, d;
    edge_decode(t, ei, s, d);
#pragma unroll
    for (int h = 0; h < HHq; ++h) {
        float l = a_src[s * HHq + h] + a_dst[d * HHq + h];
        l = (l > 0.0f) ? l : 0.2f * l;                    // LeakyReLU(0.2)
        float* addr = maxb + d * HHq + h;                 // float-max via int punning
        if (l >= 0.0f) atomicMax((int*)addr, __float_as_int(l));
        else           atomicMin((unsigned int*)addr, (unsigned int)__float_as_int(l));
    }
}

// ------------------------------------------------- edge pass 2: segment sum -
__global__ __launch_bounds__(256) void gat_edge_denom(const int* __restrict__ ei,
                                                      const float* __restrict__ a_src,
                                                      const float* __restrict__ a_dst,
                                                      const float* __restrict__ maxb,
                                                      float* __restrict__ denom) {
    const int t = blockIdx.x * 256 + threadIdx.x;
    if (t >= ETOT) return;
    int s, d;
    edge_decode(t, ei, s, d);
#pragma unroll
    for (int h = 0; h < HHq; ++h) {
        float l = a_src[s * HHq + h] + a_dst[d * HHq + h];
        l = (l > 0.0f) ? l : 0.2f * l;
        atomicAddF32(denom + d * HHq + h, __expf(l - maxb[d * HHq + h]));
    }
}

// --------------------------------- edge pass 3: weighted scatter aggregate -
// 1 block per edge, wave w handles head h=w, each lane a float4 of channels
__global__ __launch_bounds__(128) void gat_edge_agg(const int* __restrict__ ei,
                                                    const float* __restrict__ a_src,
                                                    const float* __restrict__ a_dst,
                                                    const float* __restrict__ maxb,
                                                    const float* __restrict__ denom,
                                                    const float* __restrict__ X,
                                                    float* __restrict__ agg) {
    const int t = blockIdx.x;
    int s, d;
    edge_decode(t, ei, s, d);
    const int h = threadIdx.x >> 5;
    const int c = (threadIdx.x & 31) << 2;
    float l = a_src[s * HHq + h] + a_dst[d * HHq + h];
    l = (l > 0.0f) ? l : 0.2f * l;
    const float alpha = __expf(l - maxb[d * HHq + h]) / denom[d * HHq + h];
    const float4 xv = *(const float4*)(X + (size_t)s * HCq + h * CCq + c);
    float* dst = agg + (size_t)d * HCq + h * CCq + c;
    atomicAddF32(dst + 0, alpha * xv.x);
    atomicAddF32(dst + 1, alpha * xv.y);
    atomicAddF32(dst + 2, alpha * xv.z);
    atomicAddF32(dst + 3, alpha * xv.w);
}

// ------------------------------------- GEMM2: out = aggbf16 @ Wo^T + bo -----
__global__ __launch_bounds__(128) void gat_gemm2(const __bf16* __restrict__ AGG,
                                                 const __bf16* __restrict__ WO,
                                                 const float* __restrict__ BO,
                                                 float* __restrict__ OUT) {
    const int lane = threadIdx.x & 31;
    const int wave = threadIdx.x >> 5;
    const int lo   = lane & 15;
    const int hi   = lane >> 4;
    const int n0   = blockIdx.x << 4;
    const int m0   = (blockIdx.y << 6) + (wave << 4);
    const __bf16* __restrict__ arow = AGG + (size_t)(n0 + lo) * HCq;
    const __bf16* __restrict__ wrow = WO + (size_t)(m0 + lo) * HCq;
    v8f acc = {};
#pragma unroll 4
    for (int kk = 0; kk < HCq; kk += 32) {
        v16bf afrag = load_afrag(arow, kk + (hi << 4));
        v16bf bfrag = *(const v16bf*)(wrow + kk + (hi << 4));
        acc = __builtin_amdgcn_wmma_f32_16x16x32_bf16(false, afrag, false, bfrag,
                                                      (short)0, acc, false, false);
    }
    const int col = m0 + lo;
    const int rb  = n0 + (hi << 3);
    const float bo = BO[col];
#pragma unroll
    for (int r = 0; r < 8; ++r)
        OUT[(size_t)(rb + r) * DDIM + col] = acc[r] + bo;
}

extern "C" void kernel_launch(void* const* d_in, const int* in_sizes, int n_in,
                              void* d_out, int out_size, void* d_ws, size_t ws_size,
                              hipStream_t stream) {
    const float* S    = (const float*)d_in[0];
    const int*   EI   = (const int*)d_in[1];
    const float* W    = (const float*)d_in[2];
    const float* ASRC = (const float*)d_in[3];
    const float* ADST = (const float*)d_in[4];
    const float* BIAS = (const float*)d_in[5];
    const float* WO   = (const float*)d_in[6];
    const float* BO   = (const float*)d_in[7];
    float* OUT = (float*)d_out;

    float* ws    = (float*)d_ws;
    float* X     = ws;                              // N*512 f32  (16 MB)
    float* AGG   = X + (size_t)NNq * HCq;           // N*512 f32  (16 MB)
    float* a_src = AGG + (size_t)NNq * HCq;         // N*H
    float* a_dst = a_src + NNq * HHq;
    float* maxb  = a_dst + NNq * HHq;
    float* denom = maxb + NNq * HHq;
    __bf16* Sbf   = (__bf16*)(denom + NNq * HHq);   // N*128 bf16 (2 MB)
    __bf16* Wbf   = Sbf + (size_t)NNq * DDIM;       // 512*128 bf16
    __bf16* WObf  = Wbf + (size_t)HCq * DDIM;       // 128*512 bf16
    __bf16* AGGbf = WObf + (size_t)DDIM * HCq;      // N*512 bf16 (8 MB)

    gat_cvt<<<(NNq * DDIM) / 2048, 256, 0, stream>>>(S, Sbf);
    gat_cvt<<<(HCq * DDIM) / 2048, 256, 0, stream>>>(W, Wbf);
    gat_cvt<<<(DDIM * HCq) / 2048, 256, 0, stream>>>(WO, WObf);
    gat_init<<<(NNq * HCq) / 256, 256, 0, stream>>>(AGG, maxb, denom);
    gat_gemm1<<<dim3(NNq / 16, HCq / 64), 128, 0, stream>>>(Sbf, Wbf, X);
    gat_attn_terms<<<(NNq * HHq) / 8, 256, 0, stream>>>(X, ASRC, ADST, a_src, a_dst);
    gat_edge_max<<<(ETOT + 255) / 256, 256, 0, stream>>>(EI, a_src, a_dst, maxb);
    gat_edge_denom<<<(ETOT + 255) / 256, 256, 0, stream>>>(EI, a_src, a_dst, maxb, denom);
    gat_edge_agg<<<ETOT, 128, 0, stream>>>(EI, a_src, a_dst, maxb, denom, X, AGG);
    gat_cvt_agg<<<(NNq * HCq) / 2048, 256, 0, stream>>>(AGG, BIAS, AGGbf);
    gat_gemm2<<<dim3(NNq / 16, DDIM / 64), 128, 0, stream>>>(AGGbf, WObf, BO, OUT);
}
